// SelfAttention_61340722921880
// MI455X (gfx1250) — matrix operand order repaired and checked by reading, not compiled, not run on hardware
//
#include <hip/hip_runtime.h>
#include <hip/hip_bf16.h>

// ---------------------------------------------------------------------------
// MHA forward for MI455X (gfx1250, wave32, WMMA f32_16x16x32_f16).
// Pipeline: f32->f16 convert | QKV gemms (f16 in, f32 acc, f16 out, head-split
// + V transposed) | flash attention (online softmax, WMMA for QK^T and PV) |
// output projection (f32 out + bias).
// ---------------------------------------------------------------------------

typedef __attribute__((ext_vector_type(16))) _Float16 v16h;
typedef __attribute__((ext_vector_type(8)))  _Float16 v8h;
typedef __attribute__((ext_vector_type(8)))  float    v8f;

#define D_MODEL 1024
#define NHEAD   16
#define HD      64
#define SEQ     2048
#define BATCH   2
#define MROWS   (BATCH * SEQ)      // 4096 token rows
#define LOG2E   1.4426950408889634f

static __device__ __forceinline__ v8f wmma16(v16h a, v16h b, v8f c) {
  // v_wmma_f32_16x16x32_f16 : D = A(16x32,f16) * B(32x16,f16) + C(16x16,f32)
  return __builtin_amdgcn_wmma_f32_16x16x32_f16(
      /*neg_a=*/false, a, /*neg_b=*/false, b,
      /*c_mod=*/(short)0, c, /*reuse_a=*/false, /*reuse_b=*/false);
}

// A-fragment (16x32 f16, row-major source, K contiguous).
// lane L: row = L&15 ; hi = L>>4 selects K = hi*16 .. +15 (contiguous).
static __device__ __forceinline__ v16h ldA(const _Float16* p, int hi) {
  union { v16h v; v8h h[2]; } u;
  u.h[0] = *(const v8h*)(p + hi * 16);
  u.h[1] = *(const v8h*)(p + hi * 16 + 8);
  return u.v;
}

// B-fragment (32x16 f16). lane L: col n = L&15, holds K = (L>>4)*16 .. +15,
// contiguous in memory when source rows are contiguous along K.
// caller passes p already offset by (L>>4)*16.
static __device__ __forceinline__ v16h ldB(const _Float16* p) {
  union { v16h v; v8h h[2]; } u;
  u.h[0] = *(const v8h*)(p);
  u.h[1] = *(const v8h*)(p + 8);
  return u.v;
}

// reductions across the 16-lane halves of a wave32 (masks 1,2,4,8 keep
// lanes 0-15 and 16-31 independent -> matches C-fragment row split).
static __device__ __forceinline__ float rmax16(float v) {
#pragma unroll
  for (int m = 1; m < 16; m <<= 1) v = fmaxf(v, __shfl_xor(v, m, 32));
  return v;
}
static __device__ __forceinline__ float rsum16(float v) {
#pragma unroll
  for (int m = 1; m < 16; m <<= 1) v += __shfl_xor(v, m, 32);
  return v;
}

// ---------------------------------------------------------------------------
__global__ void conv_f32_f16(const float* __restrict__ x,
                             _Float16* __restrict__ y, int n) {
  int i = blockIdx.x * blockDim.x + threadIdx.x;
  if (i < n) y[i] = (_Float16)x[i];
}

// ---------------------------------------------------------------------------
// y = X(f16,[4096,1024]) * W^T(f16,[1024,1024]) + bias ; f16 epilogue.
// mode 0: head-split  [B,H,S,hd]   (Q and K)
// mode 1: transposed  [B,H,hd,S]   (V^T, packed 16B stores)
// block = 128 (4 waves); block tile 64Mx64N; wave tile 16Mx64N; K step 32.
__global__ void gemm_qkvt(const _Float16* __restrict__ X,
                          const _Float16* __restrict__ W,
                          const float* __restrict__ bias,
                          _Float16* __restrict__ out, int mode) {
  const int wv   = threadIdx.x >> 5;
  const int lane = threadIdx.x & 31;
  const int lidx = lane & 15;
  const int hi   = lane >> 4;
  const int m0   = blockIdx.x * 64 + wv * 16;
  const int n0   = blockIdx.y * 64;

  v8f c[4] = {v8f{}, v8f{}, v8f{}, v8f{}};
  const _Float16* xrow = X + (size_t)(m0 + lidx) * D_MODEL;

#pragma unroll 1
  for (int k = 0; k < D_MODEL; k += 32) {
    v16h a = ldA(xrow + k, hi);
#pragma unroll
    for (int j = 0; j < 4; ++j) {
      v16h b = ldB(W + (size_t)(n0 + j * 16 + lidx) * D_MODEL + k + hi * 16);
      c[j] = wmma16(a, b, c[j]);
    }
  }

#pragma unroll
  for (int j = 0; j < 4; ++j) {
    const int col = n0 + j * 16 + lidx;
    const float bv = bias[col];
    const int hh = col >> 6, dd = col & 63;
    if (mode == 0) {
#pragma unroll
      for (int r = 0; r < 8; ++r) {
        const int row = m0 + hi * 8 + r;          // b*SEQ + s
        const int bb = row >> 11, ss = row & (SEQ - 1);
        out[(((size_t)bb * NHEAD + hh) * SEQ + ss) * HD + dd] =
            (_Float16)(c[j][r] + bv);
      }
    } else {
      union { v8h v; _Float16 e[8]; } t;
#pragma unroll
      for (int r = 0; r < 8; ++r) t.e[r] = (_Float16)(c[j][r] + bv);
      const int row0 = m0 + hi * 8;               // 8 consecutive s
      const int bb = row0 >> 11, ss = row0 & (SEQ - 1);
      *(v8h*)&out[(((size_t)bb * NHEAD + hh) * HD + dd) * SEQ + ss] = t.v;
    }
  }
}

// ---------------------------------------------------------------------------
// Flash attention. grid = (SEQ/64, BATCH*NHEAD); block = 128 (4 waves).
// Each wave: 16 queries x hd 64, streaming keys in tiles of 32.
__global__ void flash_attn(const _Float16* __restrict__ Q,   // [B,H,S,hd]
                           const _Float16* __restrict__ K,   // [B,H,S,hd]
                           const _Float16* __restrict__ VT,  // [B,H,hd,S]
                           _Float16* __restrict__ AO) {      // [B*S, D]
  __shared__ _Float16 plds[4 * 16 * 32];   // per-wave P-transpose staging

  const int wv   = threadIdx.x >> 5;
  const int lane = threadIdx.x & 31;
  const int lidx = lane & 15;
  const int hi   = lane >> 4;
  const int bh   = blockIdx.y;              // b*NHEAD + h
  const int b    = bh >> 4;
  const int h    = bh & (NHEAD - 1);
  const int q0   = blockIdx.x * 64 + wv * 16;

  const _Float16* Qb  = Q  + ((size_t)bh * SEQ + q0) * HD;
  const _Float16* Kb  = K  + (size_t)bh * SEQ * HD;
  const _Float16* VTb = VT + (size_t)bh * HD * SEQ;
  _Float16* pl = plds + wv * 512;

  // Q fragments for this wave (hd 0..31 and 32..63)
  const _Float16* qrow = Qb + (size_t)lidx * HD;
  const v16h aq0 = ldA(qrow, hi);
  const v16h aq1 = ldA(qrow + 32, hi);

  const float SCALE = 0.125f;               // 1/sqrt(64)
  float m[8], l[8];
  v8f o[4] = {v8f{}, v8f{}, v8f{}, v8f{}};
#pragma unroll
  for (int r = 0; r < 8; ++r) { m[r] = -1.0e30f; l[r] = 0.0f; }

#pragma unroll 1
  for (int kt = 0; kt < SEQ / 32; ++kt) {
    const int kb = kt * 32;
    if (kt + 1 < SEQ / 32) {                 // prefetch next K tile
      __builtin_prefetch(Kb + (size_t)(kb + 32 + lane) * HD, 0, 1);
    }
    // ---- scores: S = Q * K^T (16 queries x 32 keys) -----------------------
    v8f c01[2];
#pragma unroll
    for (int t = 0; t < 2; ++t) {
      const _Float16* kp = Kb + (size_t)(kb + 16 * t + lidx) * HD + hi * 16;
      v8f c = {};
      c = wmma16(aq0, ldB(kp), c);           // hd 0..31
      c = wmma16(aq1, ldB(kp + 32), c);      // hd 32..63
      c01[t] = c;
    }
    // ---- online softmax ---------------------------------------------------
#pragma unroll
    for (int r = 0; r < 8; ++r) {
      const float s0 = c01[0][r] * SCALE;
      const float s1 = c01[1][r] * SCALE;
      const float mn = fmaxf(m[r], rmax16(fmaxf(s0, s1)));
      const float alpha = exp2f((m[r] - mn) * LOG2E);
      const float p0 = exp2f((s0 - mn) * LOG2E);
      const float p1 = exp2f((s1 - mn) * LOG2E);
      l[r] = l[r] * alpha + rsum16(p0 + p1);
      m[r] = mn;
#pragma unroll
      for (int j = 0; j < 4; ++j) o[j][r] *= alpha;
      const int row = hi * 8 + r;            // C-layout -> LDS (transpose)
      pl[row * 32 + lidx]      = (_Float16)p0;
      pl[row * 32 + 16 + lidx] = (_Float16)p1;
    }
    // ---- P (A-fragment via LDS) ; O += P * V ------------------------------
    // DS ops are in-order within a wave; region is wave-private -> no barrier.
    v16h pa;
    {
      union { v16h v; v8h q[2]; } u;
      const _Float16* pr = pl + lidx * 32;
      u.q[0] = *(const v8h*)(pr + hi * 16);
      u.q[1] = *(const v8h*)(pr + hi * 16 + 8);
      pa = u.v;
    }
#pragma unroll
    for (int j = 0; j < 4; ++j) {
      const _Float16* vp = VTb + (size_t)(j * 16 + lidx) * SEQ + kb + hi * 16;
      o[j] = wmma16(pa, ldB(vp), o[j]);
    }
  }

  // ---- normalize + store [B*S, D] f16 -------------------------------------
#pragma unroll
  for (int j = 0; j < 4; ++j) {
    const int col = h * HD + j * 16 + lidx;
#pragma unroll
    for (int r = 0; r < 8; ++r) {
      const int qrow_i = q0 + hi * 8 + r;
      AO[((size_t)b * SEQ + qrow_i) * D_MODEL + col] =
          (_Float16)(o[j][r] / l[r]);
    }
  }
}

// ---------------------------------------------------------------------------
// out(f32) = AO(f16,[4096,1024]) * Wo^T(f16) + bo(f32)
__global__ void gemm_out(const _Float16* __restrict__ X,
                         const _Float16* __restrict__ W,
                         const float* __restrict__ bias,
                         float* __restrict__ Y) {
  const int wv   = threadIdx.x >> 5;
  const int lane = threadIdx.x & 31;
  const int lidx = lane & 15;
  const int hi   = lane >> 4;
  const int m0   = blockIdx.x * 64 + wv * 16;
  const int n0   = blockIdx.y * 64;

  v8f c[4] = {v8f{}, v8f{}, v8f{}, v8f{}};
  const _Float16* xrow = X + (size_t)(m0 + lidx) * D_MODEL;

#pragma unroll 1
  for (int k = 0; k < D_MODEL; k += 32) {
    v16h a = ldA(xrow + k, hi);
#pragma unroll
    for (int j = 0; j < 4; ++j) {
      v16h bfrag = ldB(W + (size_t)(n0 + j * 16 + lidx) * D_MODEL + k + hi * 16);
      c[j] = wmma16(a, bfrag, c[j]);
    }
  }

#pragma unroll
  for (int j = 0; j < 4; ++j) {
    const int col = n0 + j * 16 + lidx;
    const float bv = bias[col];
#pragma unroll
    for (int r = 0; r < 8; ++r) {
      const int row = m0 + hi * 8 + r;
      Y[(size_t)row * D_MODEL + col] = c[j][r] + bv;
    }
  }
}

// ---------------------------------------------------------------------------
extern "C" void kernel_launch(void* const* d_in, const int* in_sizes, int n_in,
                              void* d_out, int out_size, void* d_ws,
                              size_t ws_size, hipStream_t stream) {
  const float* x  = (const float*)d_in[0];
  const float* Wq = (const float*)d_in[1];
  const float* bq = (const float*)d_in[2];
  const float* Wk = (const float*)d_in[3];
  const float* bk = (const float*)d_in[4];
  const float* Wv = (const float*)d_in[5];
  const float* bv = (const float*)d_in[6];
  const float* Wo = (const float*)d_in[7];
  const float* bo = (const float*)d_in[8];

  char* ws = (char*)d_ws;
  size_t off = 0;
  const size_t XB = (size_t)MROWS * D_MODEL * 2;     // 8 MB f16
  const size_t WB = (size_t)D_MODEL * D_MODEL * 2;   // 2 MB f16
  _Float16* xh  = (_Float16*)(ws + off); off += XB;
  _Float16* wqh = (_Float16*)(ws + off); off += WB;
  _Float16* wkh = (_Float16*)(ws + off); off += WB;
  _Float16* wvh = (_Float16*)(ws + off); off += WB;
  _Float16* woh = (_Float16*)(ws + off); off += WB;
  _Float16* Qh  = (_Float16*)(ws + off); off += XB;  // [B,H,S,hd]
  _Float16* Kh  = (_Float16*)(ws + off); off += XB;  // [B,H,S,hd]
  _Float16* VTh = (_Float16*)(ws + off); off += XB;  // [B,H,hd,S]
  _Float16* AOh = (_Float16*)(ws + off); off += XB;  // [B*S, D]

  const int NX = MROWS * D_MODEL;          // 4194304
  const int NW = D_MODEL * D_MODEL;        // 1048576
  conv_f32_f16<<<(NX + 255) / 256, 256, 0, stream>>>(x,  xh,  NX);
  conv_f32_f16<<<(NW + 255) / 256, 256, 0, stream>>>(Wq, wqh, NW);
  conv_f32_f16<<<(NW + 255) / 256, 256, 0, stream>>>(Wk, wkh, NW);
  conv_f32_f16<<<(NW + 255) / 256, 256, 0, stream>>>(Wv, wvh, NW);
  conv_f32_f16<<<(NW + 255) / 256, 256, 0, stream>>>(Wo, woh, NW);

  dim3 gg(MROWS / 64, D_MODEL / 64);       // (64,16)
  gemm_qkvt<<<gg, 128, 0, stream>>>(xh, wqh, bq, Qh,  0);
  gemm_qkvt<<<gg, 128, 0, stream>>>(xh, wkh, bk, Kh,  0);
  gemm_qkvt<<<gg, 128, 0, stream>>>(xh, wvh, bv, VTh, 1);

  flash_attn<<<dim3(SEQ / 64, BATCH * NHEAD), 128, 0, stream>>>(Qh, Kh, VTh, AOh);

  gemm_out<<<gg, 128, 0, stream>>>(AOh, woh, bo, (float*)d_out);
}
